// MultiQueryAttention_31808527794910
// MI455X (gfx1250) — matrix operand order repaired and checked by reading, not compiled, not run on hardware
//
#include <hip/hip_runtime.h>
#include <hip/hip_bf16.h>

// ---------------------------------------------------------------------------
// MI455X (gfx1250) multi-query attention, bf16 WMMA pipeline.
//   x:(2,2048,1024) f32 -> Q = xWq+bq (bf16), KV = xWkv+bkv (bf16)
//   flash attention per head (head_dim=64, single KV head), K/V chunks staged
//   into LDS by the Tensor Data Mover (TENSOR_LOAD_TO_LDS + s_wait_tensorcnt)
//   out = O*Wo + bo (f32)
// All matmuls use v_wmma_f32_16x16x32_bf16 (wave32, 16x16 tiles, K=32).
// ---------------------------------------------------------------------------

typedef __attribute__((ext_vector_type(16))) __bf16 v16bf;
typedef __attribute__((ext_vector_type(8)))  float  v8f;
typedef __attribute__((ext_vector_type(4)))  unsigned int u32x4;
typedef __attribute__((ext_vector_type(8)))  int   i32x8;
typedef __attribute__((ext_vector_type(4)))  int   i32x4;

union ABu {
    v16bf v;
    uint4 q[2];
    unsigned short s[16];
};

__device__ __forceinline__ unsigned short f32_to_bf16u(float f) {
    union { float f; unsigned int u; } c; c.f = f;
    unsigned int u = c.u;
    // round-to-nearest-even
    unsigned int r = (u + 0x7FFFu + ((u >> 16) & 1u)) >> 16;
    return (unsigned short)r;
}

__device__ __forceinline__ v8f wmma_bf16(const ABu& a, const ABu& b, v8f c) {
    // (neg_a, A, neg_b, B, c_mod, C, reuse_a, reuse_b)
    return __builtin_amdgcn_wmma_f32_16x16x32_bf16(
        false, a.v, false, b.v, (short)0, c, false, false);
}

// ---------------------------------------------------------------------------
// Generic tiled GEMM:  C[M,N] = A[M,K] * B[K,N] + bias[N]
//  - A: f32 or bf16 (row-major), converted to bf16 while staged in LDS
//  - B: f32 row-major, staged TRANSPOSED in LDS so WMMA B-operand reads are
//       two contiguous 8-u16 runs per lane (k at hf*8 and 16+hf*8 in the lane's column)
//  - 128 threads = 4 waves; WG tile 64x64; wave tile 16x64 (4 accumulators)
// ---------------------------------------------------------------------------
template <bool A_F32, bool OUT_F32>
__global__ __launch_bounds__(128)
void wmma_gemm_kernel(const void* __restrict__ Aptr,
                      const float* __restrict__ Bptr,
                      const float* __restrict__ bias,
                      void* __restrict__ Cptr,
                      int N, int K) {
    __shared__ __align__(16) unsigned short lA[64][40];   // [row][k]  (pad 8)
    __shared__ __align__(16) unsigned short lBt[64][40];  // [col][k]  (pad 8)

    const int tid  = threadIdx.x;
    const int lane = tid & 31;
    const int wave = tid >> 5;
    const int lm   = lane & 15;   // N / M index within 16-tile
    const int hf   = lane >> 4;   // lane half selects K/M sub-block

    const int m0 = blockIdx.y * 64;
    const int n0 = blockIdx.x * 64;

    v8f z = {};
    v8f acc[4] = {z, z, z, z};

    for (int k0 = 0; k0 < K; k0 += 32) {
        // ---- stage A chunk: 64 rows x 32 k ----
        {
            const int row = tid >> 1;
            const int seg = (tid & 1) * 16;
            if (A_F32) {
                const float* ag = (const float*)Aptr + (size_t)(m0 + row) * K + k0 + seg;
#pragma unroll
                for (int j = 0; j < 16; ++j) lA[row][seg + j] = f32_to_bf16u(ag[j]);
                if (k0 + 32 < K) __builtin_prefetch(ag + 32, 0, 1);
            } else {
                const unsigned short* ag =
                    (const unsigned short*)Aptr + (size_t)(m0 + row) * K + k0 + seg;
#pragma unroll
                for (int j = 0; j < 16; ++j) lA[row][seg + j] = ag[j];
                if (k0 + 32 < K) __builtin_prefetch(ag + 32, 0, 1);
            }
        }
        // ---- stage B chunk transposed: 32 k x 64 cols -> lBt[col][k] ----
        {
            const int kr = tid >> 2;
            const int cs = (tid & 3) * 16;
            const float* bg = Bptr + (size_t)(k0 + kr) * N + n0 + cs;
#pragma unroll
            for (int j = 0; j < 16; ++j) lBt[cs + j][kr] = f32_to_bf16u(bg[j]);
            if (k0 + 32 < K) __builtin_prefetch(bg + (size_t)32 * N, 0, 1);
        }
        __syncthreads();

        // ---- A operand (16x32): lane row = wave*16+lm; k segs at hf*8 and 16+hf*8
        ABu a;
        {
            const unsigned short* ar = &lA[wave * 16 + lm][0];
            a.q[0] = *(const uint4*)(ar + hf * 8);
            a.q[1] = *(const uint4*)(ar + 16 + hf * 8);
        }
        // ---- 4 B operands + WMMA ----
#pragma unroll
        for (int nt = 0; nt < 4; ++nt) {
            ABu b;
            const unsigned short* br = &lBt[nt * 16 + lm][0] + hf * 8;
            b.q[0] = *(const uint4*)(br);
            b.q[1] = *(const uint4*)(br + 16);
            acc[nt] = wmma_bf16(a, b, acc[nt]);
        }
        __syncthreads();
    }

    // ---- epilogue: bias + store (C layout: lane=N col, vgpr r -> row hf*8+r)
#pragma unroll
    for (int nt = 0; nt < 4; ++nt) {
        const int col = n0 + nt * 16 + lm;
        const float bv = bias[col];
#pragma unroll
        for (int r = 0; r < 8; ++r) {
            const int row = m0 + wave * 16 + hf * 8 + r;
            const float v = acc[nt][r] + bv;
            if (OUT_F32)
                ((float*)Cptr)[(size_t)row * N + col] = v;
            else
                ((unsigned short*)Cptr)[(size_t)row * N + col] = f32_to_bf16u(v);
        }
    }
}

// ---------------------------------------------------------------------------
// Flash attention (multi-query: one shared K/V head of dim 64).
//   Q : bf16 [4096][1024] row = b*2048+n, col = h*64+d
//   KV: bf16 [4096][128]  cols 0..63 = k, 64..127 = v
//   O : bf16 [4096][1024]
// Grid (32, 16, 2) = (query blocks of 64, heads, batch); 128 threads = 4 waves.
//
// Per 32-key chunk the whole 32x128 bf16 KV tile is DMA'd into LDS by the
// Tensor Data Mover (wave 0 issues TENSOR_LOAD_TO_LDS, waits TENSORcnt, then
// the workgroup barrier publishes it).  TDM pad fields insert 4 dwords of LDS
// padding per 64-dword row (256B -> 272B stride) so per-key-row reads hit
// distinct LDS banks.
// ---------------------------------------------------------------------------

// padded LDS row stride for the KV chunk, in u16 units: 128 data + 8 pad
#define KV_ROW 136

__global__ __launch_bounds__(128)
void mqa_flash_kernel(const unsigned short* __restrict__ Q,
                      const unsigned short* __restrict__ KV,
                      unsigned short* __restrict__ O) {
    __shared__ __align__(16) unsigned short lKV[32 * KV_ROW];  // [key][128 d + pad]
    __shared__ __align__(16) unsigned short lP[4][16][40];     // per-wave [qrow][key]

    const int tid  = threadIdx.x;
    const int lane = tid & 31;
    const int wave = tid >> 5;
    const int lm   = lane & 15;
    const int hf   = lane >> 4;

    const int b  = blockIdx.z;
    const int h  = blockIdx.y;
    const int m0 = blockIdx.x * 64;

    // --- load Q tile (16 rows x 64 d) into two A operands, kept in VGPRs ---
    const int qrow = b * 2048 + m0 + wave * 16 + lm;
    const unsigned short* Qp = Q + (size_t)qrow * 1024 + h * 64;
    ABu aq0, aq1;
    aq0.q[0] = *(const uint4*)(Qp + hf * 8);
    aq0.q[1] = *(const uint4*)(Qp + 16 + hf * 8);
    aq1.q[0] = *(const uint4*)(Qp + 32 + hf * 8);
    aq1.q[1] = *(const uint4*)(Qp + 48 + hf * 8);

    v8f z = {};
    v8f acc[4] = {z, z, z, z};
    float mrun[8], lsum[8];
#pragma unroll
    for (int r = 0; r < 8; ++r) { mrun[r] = -3.0e38f; lsum[r] = 0.0f; }

    const float scale = 0.125f;  // 1/sqrt(64)

    for (int kb = 0; kb < 2048; kb += 32) {
        // ================= stage KV chunk (32 keys x 128 bf16) ==============
#if __has_builtin(__builtin_amdgcn_tensor_load_to_lds)
        if (wave == 0) {
            // ---- Tensor DMA Descriptor (D#), ISA 8.3/8.4 bit layout ----
            const unsigned long long ga =
                (unsigned long long)(const void*)(KV + (size_t)(b * 2048 + kb) * 128);
            const unsigned lds = (unsigned)(unsigned long long)(const void*)&lKV[0];
            u32x4 g0 = {
                1u,                                       // count=1 (user D#)
                lds,                                      // lds_addr (bytes)
                (unsigned)ga,                             // global_addr[31:0]
                (unsigned)((ga >> 32) & 0x1FFFFFFull)     // global_addr[56:32]
                    | (2u << 30)                          // type=2 ("image")
            };
            i32x8 g1 = {
                (int)((1u << 16)      // data_size = 1 -> 2 bytes
                    | (1u << 20)      // pad_enable
                    | (5u << 22)      // pad_interval: every 64 dwords (256B row)
                    | (3u << 25)),    // pad_amount: 4 dwords (16B) per row
                (int)(128u << 16),    // tensor_dim0 = 128 elements/row
                (int)(2048u << 16),   // tensor_dim1 = 2048 rows
                (int)(128u << 16),    // tile_dim0 = 128
                32,                   // tile_dim1 = 32 keys
                128,                  // tensor_dim0_stride = 128 elements
                0, 0                  // tensor_dim1_stride unused (2D)
            };
            i32x4 gz = {0, 0, 0, 0};  // groups 2/3 unused for 2D tensors
#if __has_include(<hip/amd_detail/amd_gfx1250_TDM.h>)
            i32x8 gz8 = {0, 0, 0, 0, 0, 0, 0, 0};
            __builtin_amdgcn_tensor_load_to_lds(g0, g1, gz, gz, gz8, 0);
#else
            __builtin_amdgcn_tensor_load_to_lds(g0, g1, gz, gz, 0);
#endif
            __builtin_amdgcn_s_wait_tensorcnt(0);
        }
#else
        // fallback: cooperative vector staging (same padded layout)
        {
            const int kr = tid >> 2;          // key 0..31
            const int cs = (tid & 3) * 32;    // 32 cols per thread
            const unsigned short* kvp = KV + (size_t)(b * 2048 + kb + kr) * 128 + cs;
#pragma unroll
            for (int j = 0; j < 32; ++j) lKV[kr * KV_ROW + cs + j] = kvp[j];
        }
#endif
        __syncthreads();

        // ---- S = Q K^T for two 16-key tiles (accumulate over d in 2 steps) ----
        v8f s0 = z, s1 = z;
        {
            ABu bk;
            const unsigned short* kr0 = lKV + lm * KV_ROW;         // tile0 key = lm
            bk.q[0] = *(const uint4*)(kr0 + hf * 8);
            bk.q[1] = *(const uint4*)(kr0 + 16 + hf * 8);
            s0 = wmma_bf16(aq0, bk, s0);
            bk.q[0] = *(const uint4*)(kr0 + 32 + hf * 8);
            bk.q[1] = *(const uint4*)(kr0 + 48 + hf * 8);
            s0 = wmma_bf16(aq1, bk, s0);

            const unsigned short* kr1 = lKV + (16 + lm) * KV_ROW;  // tile1 key
            bk.q[0] = *(const uint4*)(kr1 + hf * 8);
            bk.q[1] = *(const uint4*)(kr1 + 16 + hf * 8);
            s1 = wmma_bf16(aq0, bk, s1);
            bk.q[0] = *(const uint4*)(kr1 + 32 + hf * 8);
            bk.q[1] = *(const uint4*)(kr1 + 48 + hf * 8);
            s1 = wmma_bf16(aq1, bk, s1);
        }

        // ---- streaming softmax (rows = hf*8+r; reduce over 16-lane half) ----
        float mnew[8], alpha[8], p0[8], p1[8];
#pragma unroll
        for (int r = 0; r < 8; ++r) {
            float a0 = s0[r] * scale, a1 = s1[r] * scale;
            s0[r] = a0; s1[r] = a1;
            float t = fmaxf(a0, a1);
            t = fmaxf(t, __shfl_xor(t, 1));
            t = fmaxf(t, __shfl_xor(t, 2));
            t = fmaxf(t, __shfl_xor(t, 4));
            t = fmaxf(t, __shfl_xor(t, 8));
            mnew[r]  = fmaxf(mrun[r], t);
            alpha[r] = __expf(mrun[r] - mnew[r]);
            mrun[r]  = mnew[r];
        }
#pragma unroll
        for (int r = 0; r < 8; ++r) {
            p0[r] = __expf(s0[r] - mnew[r]);
            p1[r] = __expf(s1[r] - mnew[r]);
            float t = p0[r] + p1[r];
            t += __shfl_xor(t, 1);
            t += __shfl_xor(t, 2);
            t += __shfl_xor(t, 4);
            t += __shfl_xor(t, 8);
            lsum[r] = lsum[r] * alpha[r] + t;
        }
#pragma unroll
        for (int nt = 0; nt < 4; ++nt)
#pragma unroll
            for (int r = 0; r < 8; ++r) acc[nt][r] *= alpha[r];

        // ---- P: C-layout -> LDS -> A-operand layout (per-wave scratch) ----
#pragma unroll
        for (int r = 0; r < 8; ++r) {
            lP[wave][hf * 8 + r][lm]      = f32_to_bf16u(p0[r]);
            lP[wave][hf * 8 + r][16 + lm] = f32_to_bf16u(p1[r]);
        }
        // same-wave LDS RAW: stores and loads are in-order per wave; the wait +
        // memory clobber stops compiler reordering across the transpose.
        asm volatile("s_wait_dscnt 0x0" ::: "memory");
        ABu ap;
        {
            const unsigned short* pr = &lP[wave][lm][0];
            ap.q[0] = *(const uint4*)(pr + hf * 8);
            ap.q[1] = *(const uint4*)(pr + 16 + hf * 8);
        }

        // ---- O += P * V: B operand gathers key-strided u16 from the padded
        //      row-major KV tile (16 distinct banks thanks to TDM padding) ----
#pragma unroll
        for (int nt = 0; nt < 4; ++nt) {
            ABu bv;
            const int col = 64 + nt * 16 + lm;   // V columns are 64..127
#pragma unroll
            for (int j = 0; j < 16; ++j)
                bv.s[j] = lKV[(hf * 8 + j + (j >= 8 ? 8 : 0)) * KV_ROW + col];
            acc[nt] = wmma_bf16(ap, bv, acc[nt]);
        }
        __syncthreads();
    }

    // ---- normalize and store O (bf16) ----
#pragma unroll
    for (int nt = 0; nt < 4; ++nt) {
        const int col = h * 64 + nt * 16 + lm;
#pragma unroll
        for (int r = 0; r < 8; ++r) {
            const int row = b * 2048 + m0 + wave * 16 + hf * 8 + r;
            O[(size_t)row * 1024 + col] = f32_to_bf16u(acc[nt][r] / lsum[r]);
        }
    }
}

// ---------------------------------------------------------------------------
// Host launcher
// ---------------------------------------------------------------------------
extern "C" void kernel_launch(void* const* d_in, const int* in_sizes, int n_in,
                              void* d_out, int out_size, void* d_ws, size_t ws_size,
                              hipStream_t stream) {
    (void)in_sizes; (void)n_in; (void)out_size; (void)ws_size;

    const float* x   = (const float*)d_in[0];  // (2,2048,1024)
    const float* Wq  = (const float*)d_in[1];  // (1024,1024)
    const float* bq  = (const float*)d_in[2];  // (1024)
    const float* Wkv = (const float*)d_in[3];  // (1024,128)
    const float* bkv = (const float*)d_in[4];  // (128)
    const float* Wo  = (const float*)d_in[5];  // (1024,1024)
    const float* bo  = (const float*)d_in[6];  // (1024)
    float* out = (float*)d_out;                // (2,2048,1024)

    const int M = 4096;   // B*N tokens
    const int D = 1024;
    const int DKV = 128;

    // workspace: Q (8MB) | KV (1MB) | O (8MB), all bf16
    unsigned short* Qb  = (unsigned short*)d_ws;
    unsigned short* KVb = Qb + (size_t)M * D;
    unsigned short* Ob  = KVb + (size_t)M * DKV;

    dim3 blk(128);

    // 1) Q = x*Wq + bq   (f32 A -> bf16 out)
    wmma_gemm_kernel<true, false><<<dim3(D / 64, M / 64), blk, 0, stream>>>(
        (const void*)x, Wq, bq, (void*)Qb, D, D);

    // 2) KV = x*Wkv + bkv
    wmma_gemm_kernel<true, false><<<dim3(DKV / 64, M / 64), blk, 0, stream>>>(
        (const void*)x, Wkv, bkv, (void*)KVb, DKV, D);

    // 3) flash attention: grid (qblocks, heads, batch)
    mqa_flash_kernel<<<dim3(2048 / 64, 16, 2), blk, 0, stream>>>(Qb, KVb, Ob);

    // 4) out = O*Wo + bo  (bf16 A -> f32 out)
    wmma_gemm_kernel<false, true><<<dim3(D / 64, M / 64), blk, 0, stream>>>(
        (const void*)Ob, Wo, bo, (void*)out, D, D);
}
